// Model_NoGNN_4526895530527
// MI455X (gfx1250) — matrix operand order repaired and checked by reading, not compiled, not run on hardware
//
#include <hip/hip_runtime.h>
#include <hip/hip_bf16.h>
#include <math.h>

typedef _Float16 h16;
typedef __attribute__((ext_vector_type(16))) _Float16 v16h;
typedef __attribute__((ext_vector_type(8)))  float    v8f;
typedef unsigned int u32x4 __attribute__((ext_vector_type(4)));
typedef int          i32x4 __attribute__((ext_vector_type(4)));
typedef int          i32x8 __attribute__((ext_vector_type(8)));

#define BB    32
#define NN    256
#define LL    1024
#define DIMM  128
#define MODES 8
#define HH    4
#define DHH   32
#define FFD   256

// ---------------------------------------------------------------------------
// WMMA helper: D = A(16x32 f16) * B(32x16 f16) + C(16x16 f32)
// ---------------------------------------------------------------------------
__device__ inline v8f wmma32(v16h a, v16h b, v8f c) {
  return __builtin_amdgcn_wmma_f32_16x16x32_f16(
      /*neg_a=*/false, a, /*neg_b=*/false, b,
      /*c_mod=*/(short)0, c, /*reuse_a=*/false, /*reuse_b=*/false);
}

// Cross-lane reductions (wave32). Rows of a WMMA C tile live entirely within
// one 16-lane half, so masks 1..8 reduce a row; masks 1..16 reduce the wave.
__device__ inline float red_max16(float x) {
#pragma unroll
  for (int m = 1; m < 16; m <<= 1) x = fmaxf(x, __shfl_xor(x, m, 32));
  return x;
}
__device__ inline float red_sum16(float x) {
#pragma unroll
  for (int m = 1; m < 16; m <<= 1) x += __shfl_xor(x, m, 32);
  return x;
}
__device__ inline float red_sum32(float x) {
#pragma unroll
  for (int m = 1; m < 32; m <<= 1) x += __shfl_xor(x, m, 32);
  return x;
}
__device__ inline float red_max32(float x) {
#pragma unroll
  for (int m = 1; m < 32; m <<= 1) x = fmaxf(x, __shfl_xor(x, m, 32));
  return x;
}

// ---------------------------------------------------------------------------
// TDM: stage a 2D weight panel [tile_rows x K] (f16, row-major, stride K)
// from global into LDS. Builds the D# per CDNA5 ISA (group0/group1
// bitfields), issues tensor_load_to_lds, tracked by TENSORcnt.
// ---------------------------------------------------------------------------
#if __has_builtin(__builtin_amdgcn_tensor_load_to_lds)
#define HAVE_TDM 1
__device__ inline void tdm_load_panel(unsigned int lds_off, const h16* gsrc,
                                      unsigned int K, unsigned int nrows,
                                      unsigned int tile_rows) {
  unsigned long long ga = (unsigned long long)(uintptr_t)gsrc;
  u32x4 g0;
  g0[0] = 1u;                                   // count=1, user mode
  g0[1] = lds_off;                              // lds_addr  (bits 63:32)
  g0[2] = (unsigned int)ga;                     // global_addr[31:0]
  g0[3] = (unsigned int)((ga >> 32) & 0x1FFFFFFu) | (2u << 30);  // addr[56:32] | type=2
  i32x8 g1;
  g1[0] = (int)(1u << 16);                      // data_size=1 (2 bytes)
  g1[1] = (int)((K & 0xFFFFu) << 16);           // tensor_dim0 lo16 (bits 63:48)
  g1[2] = (int)((K >> 16) | ((nrows & 0xFFFFu) << 16));   // dim0 hi | dim1 lo
  g1[3] = (int)((nrows >> 16) | (K << 16));     // dim1 hi | tile_dim0 = K
  g1[4] = (int)tile_rows;                       // tile_dim1
  g1[5] = (int)K;                               // tensor_dim0_stride lo32
  g1[6] = 0;
  g1[7] = 0;
  i32x4 g2 = {};  // unused (2D tensor)
  i32x4 g3 = {};
#if __has_include(<hip/amd_detail/amd_gfx1250_TDM.h>)
  // amdgpu-toolchain (clang-23 / therock): 6-arg form
  i32x8 g4 = {};
  __builtin_amdgcn_tensor_load_to_lds(g0, g1, g2, g3, g4, 0);
#else
  // ROCm 7.2 (clang-22): 5-arg form
  __builtin_amdgcn_tensor_load_to_lds(g0, g1, g2, g3, 0);
#endif
}
#else
#define HAVE_TDM 0
#endif

// ---------------------------------------------------------------------------
// f32 -> f16 convert (weights, acc_Y). npad >= n: zero-pads the tail so the
// N=8 weight matrices become clean 16-row panels.
// ---------------------------------------------------------------------------
__global__ void k_f32_to_f16(const float* __restrict__ in, h16* __restrict__ out,
                             int n, int npad) {
  int i = blockIdx.x * blockDim.x + threadIdx.x;
  if (i < npad) out[i] = (i < n) ? (h16)in[i] : (h16)0.f;
}

// ---------------------------------------------------------------------------
// proj_in: x[b,l,d] = sum_j accY[b,2+j,l] * w[d,j] + b[d]   (K=7, VALU)
// ---------------------------------------------------------------------------
__global__ void k_proj_in(const float* __restrict__ accY, const float* __restrict__ w,
                          const float* __restrict__ bias, float* __restrict__ xf,
                          h16* __restrict__ xh) {
  int t = blockIdx.x * blockDim.x + threadIdx.x;  // B*L*DIM
  if (t >= BB * LL * DIMM) return;
  int d  = t & (DIMM - 1);
  int bl = t >> 7;
  int l  = bl & (LL - 1);
  int b  = bl >> 10;
  const float* wr = w + d * 7;
  float acc = bias[d];
#pragma unroll
  for (int j = 0; j < 7; ++j)
    acc += accY[((size_t)(b * NN + 2 + j)) * LL + l] * wr[j];
  xf[t] = acc;
  xh[t] = (h16)acc;
}

// ---------------------------------------------------------------------------
// Generic WMMA GEMM (fallback path for K=1024 and tiny-N cases):
// out = act(A[M,K] * W[Np,K]^T + bias [+ resid]); W is padded to Np (mult 16)
// so the B-fragment load needs no guard; only the store masks n < N.
// ---------------------------------------------------------------------------
__global__ void k_gemm(const h16* __restrict__ A, const h16* __restrict__ W,
                       const float* __restrict__ bias, const float* __restrict__ resid,
                       float* __restrict__ oF, h16* __restrict__ oH,
                       int M, int N, int Np, int K, int act) {
  int lane = threadIdx.x & 31;
  int wid  = blockIdx.x * (blockDim.x >> 5) + (threadIdx.x >> 5);
  int tn   = Np >> 4;
  int tiles = (M >> 4) * tn;
  if (wid >= tiles) return;  // wave-uniform: EXEC stays all-1s for WMMA
  int tm = wid / tn, tc = wid % tn;
  int row0 = tm << 4, col0 = tc << 4;
  int mA = row0 + (lane & 15);
  int kh = lane >> 4;               // K-half selector (A and B frags)
  int nB = col0 + (lane & 15);
  const h16* pa0 = A + (size_t)mA * K;
  const h16* pb0 = W + (size_t)nB * K + (kh << 4);
  v8f c = {};
  for (int k0 = 0; k0 < K; k0 += 32) {
    v16h a;
    const h16* pa = pa0 + k0;
#pragma unroll
    for (int v = 0; v < 8; ++v) {
      int k = (kh << 4) + (v << 1);
      a[2 * v]     = pa[k];
      a[2 * v + 1] = pa[k + 1];
    }
    v16h bf;
    const h16* pb = pb0 + k0;
#pragma unroll
    for (int r = 0; r < 8; ++r) { bf[2 * r] = pb[2 * r]; bf[2 * r + 1] = pb[2 * r + 1]; }
    if (k0 + 32 < K) __builtin_prefetch(pa + 64, 0, 0);   // global_prefetch next A tile
    c = wmma32(a, bf, c);
  }
  if (nB < N) {
    float bv = bias ? bias[nB] : 0.f;
    int rb = row0 + (kh << 3);  // C layout: comp v -> row v (+8 upper half)
#pragma unroll
    for (int v = 0; v < 8; ++v) {
      int m = rb + v;
      float x = c[v] + bv;
      if (resid) x += resid[(size_t)m * N + nB];
      if (act == 1) x = fmaxf(x, 0.f);
      else if (act == 2) x = tanhf(x);
      size_t off = (size_t)m * N + nB;
      if (oF) oF[off] = x;
      if (oH) oH[off] = (h16)x;
    }
  }
}

// ---------------------------------------------------------------------------
// LDS-staged WMMA GEMM for the heavy layers (N mult of 64, K <= 256):
// block = 8 waves = 128 rows x 64 cols. The 64xK weight panel is staged into
// LDS once per block by the Tensor Data Mover (TENSORcnt), then every k-step
// is: 1 A-frag (2x global b128) + 4 B-frags (2x ds b128 each) + 4 WMMAs.
// ---------------------------------------------------------------------------
__global__ void k_gemm_lds(const h16* __restrict__ A, const h16* __restrict__ W,
                           const float* __restrict__ bias, const float* __restrict__ resid,
                           float* __restrict__ oF, h16* __restrict__ oH,
                           int M, int N, int K, int act) {
  __shared__ h16 wpanel[64 * 256];   // 32 KB (<= 64 rows x K<=256)
  int lane = threadIdx.x & 31;
  int wave = threadIdx.x >> 5;
  int row0 = blockIdx.x * 128 + wave * 16;
  int col0 = blockIdx.y * 64;

  // stage weight panel W[col0 .. col0+63][0..K) into LDS
#if HAVE_TDM
  if (threadIdx.x < 32) {            // wave 0 issues the TDM descriptor
    tdm_load_panel((unsigned int)(uintptr_t)&wpanel[0],
                   W + (size_t)col0 * K, (unsigned int)K,
                   (unsigned int)(N - col0), 64u);
    __builtin_amdgcn_s_wait_tensorcnt(0);
  }
#else
  for (int i = threadIdx.x * 8; i < 64 * K; i += 256 * 8)
    *(ulonglong2*)&wpanel[i] = *(const ulonglong2*)&W[(size_t)col0 * K + i];
#endif
  __syncthreads();

  int mA = row0 + (lane & 15);
  int kh = lane >> 4;
  const h16* pa0 = A + (size_t)mA * K;
  v8f acc[4] = {{}, {}, {}, {}};
  for (int k0 = 0; k0 < K; k0 += 32) {
    v16h a;
    const h16* pa = pa0 + k0;
#pragma unroll
    for (int v = 0; v < 8; ++v) {
      int k = (kh << 4) + (v << 1);
      a[2 * v]     = pa[k];
      a[2 * v + 1] = pa[k + 1];
    }
#pragma unroll
    for (int s = 0; s < 4; ++s) {
      int nloc = s * 16 + (lane & 15);
      const h16* pb = &wpanel[(size_t)nloc * K + k0 + (kh << 4)];  // 32B contiguous
      v16h bf;
#pragma unroll
      for (int r = 0; r < 8; ++r) { bf[2 * r] = pb[2 * r]; bf[2 * r + 1] = pb[2 * r + 1]; }
      acc[s] = wmma32(a, bf, acc[s]);
    }
  }
  int rb = row0 + (kh << 3);
#pragma unroll
  for (int s = 0; s < 4; ++s) {
    int n = col0 + s * 16 + (lane & 15);
    float bv = bias ? bias[n] : 0.f;
#pragma unroll
    for (int v = 0; v < 8; ++v) {
      int m = rb + v;
      float x = acc[s][v] + bv;
      if (resid) x += resid[(size_t)m * N + n];
      if (act == 1) x = fmaxf(x, 0.f);
      else if (act == 2) x = tanhf(x);
      size_t off = (size_t)m * N + n;
      if (oF) oF[off] = x;
      if (oH) oH[off] = (h16)x;
    }
  }
}

// ---------------------------------------------------------------------------
// Attention: per-wave (b, h, 16-query tile). DH=32 => one WMMA per score tile.
// Two-pass online softmax; P staged through 1KB/wave LDS to re-layout C->A.
// ---------------------------------------------------------------------------
__global__ void k_attn(const h16* __restrict__ qkv, h16* __restrict__ ctx) {
  int lane  = threadIdx.x & 31;
  int wslot = threadIdx.x >> 5;
  int wid   = blockIdx.x * (blockDim.x >> 5) + wslot;
  const int NQ = LL / 16;  // 64 query tiles
  if (wid >= BB * HH * NQ) return;
  int qt = wid % NQ;
  int bh = wid / NQ;
  int h  = bh % HH;
  int b  = bh / HH;

  __shared__ h16 pst[4][16][34];  // per-wave 16x32 P staging (+pad)

  const int LDQ = 3 * DIMM;  // 384
  const h16* base = qkv + (size_t)b * LL * LDQ;

  // Q fragment (A layout, rows = queries, K = DH = 32)
  v16h qa;
  {
    int m  = (qt << 4) + (lane & 15);
    int kh = lane >> 4;
    const h16* p = base + (size_t)m * LDQ + h * DHH;
#pragma unroll
    for (int v = 0; v < 8; ++v) {
      int k = (kh << 4) + (v << 1);
      qa[2 * v]     = p[k];
      qa[2 * v + 1] = p[k + 1];
    }
  }

  // K^T B-fragment for a 16-key tile: B[kdim][key] = K[key][kdim]
  auto kfrag = [&](int kt) -> v16h {
    v16h kb;
    int key  = (kt << 4) + (lane & 15);
    int koff = (lane >> 4) << 4;
    const h16* p = base + (size_t)key * LDQ + DIMM + h * DHH + koff;
#pragma unroll
    for (int r = 0; r < 8; ++r) { kb[2 * r] = p[2 * r]; kb[2 * r + 1] = p[2 * r + 1]; }
    return kb;
  };

  const float scale = 0.1767766952966369f;  // 1/sqrt(32)
  float mrow[8], srow[8];
#pragma unroll
  for (int v = 0; v < 8; ++v) { mrow[v] = -1e30f; srow[v] = 0.f; }

  // ---- pass 1: softmax row stats (online max / sum) ----
  for (int kt = 0; kt < NQ; ++kt) {
    v8f z = {};
    v8f c = wmma32(qa, kfrag(kt), z);
#pragma unroll
    for (int v = 0; v < 8; ++v) {
      float sv   = c[v] * scale;
      float tmax = red_max16(sv);
      float nm   = fmaxf(mrow[v], tmax);
      float p    = __expf(sv - nm);
      float ps   = red_sum16(p);
      srow[v] = srow[v] * __expf(mrow[v] - nm) + ps;
      mrow[v] = nm;
    }
  }
  float rinv[8];
#pragma unroll
  for (int v = 0; v < 8; ++v) rinv[v] = 1.f / srow[v];

  // ---- pass 2: ctx = P @ V (32-key chunks; P via LDS relayout) ----
  v8f acc0 = {}, acc1 = {};
  for (int ch = 0; ch < 32; ++ch) {
#pragma unroll
    for (int sub = 0; sub < 2; ++sub) {
      v8f z = {};
      v8f c = wmma32(qa, kfrag(ch * 2 + sub), z);
      int rbase = (lane >> 4) << 3;
      int col   = lane & 15;
#pragma unroll
      for (int v = 0; v < 8; ++v) {
        float p = __expf(c[v] * scale - mrow[v]) * rinv[v];
        pst[wslot][rbase + v][(sub << 4) + col] = (h16)p;
      }
    }
    // reload P as A fragment (16 queries x 32 keys)
    v16h pa;
    {
      int m  = lane & 15;
      int kh = lane >> 4;
#pragma unroll
      for (int v = 0; v < 8; ++v) {
        int k = (kh << 4) + (v << 1);
        pa[2 * v]     = pst[wslot][m][k];
        pa[2 * v + 1] = pst[wslot][m][k + 1];
      }
    }
    // V B-fragments: B[key][dim]; dims 0..15 (acc0) and 16..31 (acc1)
    int keybase = (ch << 5) + ((lane >> 4) << 4);
    int dim     = lane & 15;
    const h16* pv = base + (size_t)keybase * LDQ + 2 * DIMM + h * DHH + dim;
    v16h vb0, vb1;
#pragma unroll
    for (int r = 0; r < 8; ++r) {
      vb0[2 * r]     = pv[(size_t)(2 * r) * LDQ];
      vb0[2 * r + 1] = pv[(size_t)(2 * r + 1) * LDQ];
      vb1[2 * r]     = pv[(size_t)(2 * r) * LDQ + 16];
      vb1[2 * r + 1] = pv[(size_t)(2 * r + 1) * LDQ + 16];
    }
    acc0 = wmma32(pa, vb0, acc0);
    acc1 = wmma32(pa, vb1, acc1);
  }

  // store ctx tile (f16), C layout
  {
    int col = lane & 15;
    int rb  = (qt << 4) + ((lane >> 4) << 3);
#pragma unroll
    for (int v = 0; v < 8; ++v) {
      h16* o = ctx + ((size_t)(b * LL + rb + v)) * DIMM + h * DHH;
      o[col]      = (h16)acc0[v];
      o[col + 16] = (h16)acc1[v];
    }
  }
}

// ---------------------------------------------------------------------------
// LayerNorm over DIM=128, one wave per row (4 elems/lane, shuffle reductions)
// ---------------------------------------------------------------------------
__global__ void k_layernorm(const float* __restrict__ in, const float* __restrict__ g,
                            const float* __restrict__ be, float* __restrict__ oF,
                            h16* __restrict__ oH, int rows) {
  int lane = threadIdx.x & 31;
  int wid  = blockIdx.x * (blockDim.x >> 5) + (threadIdx.x >> 5);
  if (wid >= rows) return;
  const float* p = in + (size_t)wid * DIMM;
  float4 xv = ((const float4*)p)[lane];
  float mean = red_sum32(xv.x + xv.y + xv.z + xv.w) * (1.f / DIMM);
  float dx = xv.x - mean, dy = xv.y - mean, dz = xv.z - mean, dw = xv.w - mean;
  float var  = red_sum32(dx * dx + dy * dy + dz * dz + dw * dw) * (1.f / DIMM);
  float rstd = rsqrtf(var + 1e-5f);
  float r[4] = {dx, dy, dz, dw};
  int i0 = lane * 4;
#pragma unroll
  for (int j = 0; j < 4; ++j) {
    float y = r[j] * rstd * g[i0 + j] + be[i0 + j];
    size_t off = (size_t)wid * DIMM + i0 + j;
    if (oF) oF[off] = y;
    if (oH) oH[off] = (h16)y;
  }
}

// ---------------------------------------------------------------------------
// phi abs-max over nodes, then normalize
// ---------------------------------------------------------------------------
__global__ void k_absmax(const float* __restrict__ phi, float* __restrict__ amax) {
  int lane = threadIdx.x & 31;
  int wid  = blockIdx.x * (blockDim.x >> 5) + (threadIdx.x >> 5);
  if (wid >= BB * MODES) return;
  int b = wid >> 3, mode = wid & 7;
  const float* p = phi + (size_t)b * NN * MODES + mode;
  float a = 0.f;
  for (int n = lane; n < NN; n += 32) a = fmaxf(a, fabsf(p[(size_t)n * MODES]));
  a = red_max32(a);
  if (lane == 0) amax[wid] = a;
}

__global__ void k_norm(const float* __restrict__ phi, const float* __restrict__ amax,
                       float* __restrict__ out) {
  int t = blockIdx.x * blockDim.x + threadIdx.x;
  if (t >= BB * NN * MODES) return;
  int mode = t & 7;
  int b    = t >> 11;  // / (NN*MODES)
  out[t] = phi[t] / amax[b * MODES + mode];
}

// ---------------------------------------------------------------------------
// Host orchestration
// ---------------------------------------------------------------------------
extern "C" void kernel_launch(void* const* d_in, const int* in_sizes, int n_in,
                              void* d_out, int out_size, void* d_ws, size_t ws_size,
                              hipStream_t stream) {
  (void)in_sizes; (void)n_in; (void)out_size; (void)ws_size;
  const float* accY       = (const float*)d_in[0];
  const float* proj_in_w  = (const float*)d_in[2];
  const float* proj_in_b  = (const float*)d_in[3];
  const float* attn_in_w  = (const float*)d_in[4];
  const float* attn_in_b  = (const float*)d_in[5];
  const float* attn_out_w = (const float*)d_in[6];
  const float* attn_out_b = (const float*)d_in[7];
  const float* ln1_g      = (const float*)d_in[8];
  const float* ln1_b      = (const float*)d_in[9];
  const float* ff1_w      = (const float*)d_in[10];
  const float* ff1_b      = (const float*)d_in[11];
  const float* ff2_w      = (const float*)d_in[12];
  const float* ff2_b      = (const float*)d_in[13];
  const float* ln2_g      = (const float*)d_in[14];
  const float* ln2_b      = (const float*)d_in[15];
  const float* proj_out_w = (const float*)d_in[16];
  const float* proj_out_b = (const float*)d_in[17];
  const float* mlp1_w     = (const float*)d_in[18];
  const float* mlp1_b     = (const float*)d_in[19];
  const float* mlp2_w     = (const float*)d_in[20];
  const float* mlp2_b     = (const float*)d_in[21];
  const float* mlp3_w     = (const float*)d_in[22];
  const float* mlp3_b     = (const float*)d_in[23];

  float* q_out   = (float*)d_out;                           // [B*L, 8]
  float* phi_out = (float*)d_out + (size_t)BB * LL * MODES; // [B*N, 8]

  const int ML = BB * LL;  // 32768 (q-path rows)
  const int MP = BB * NN;  // 8192  (phi-path rows)

  // workspace arena (256B aligned slots)
  char* cur = (char*)d_ws;
  auto alloc = [&](size_t bytes) -> void* {
    void* p = (void*)cur;
    cur += (bytes + 255) & ~(size_t)255;
    return p;
  };
  h16* wh_attn_in  = (h16*)alloc(sizeof(h16) * 3 * DIMM * DIMM);
  h16* wh_attn_out = (h16*)alloc(sizeof(h16) * DIMM * DIMM);
  h16* wh_ff1      = (h16*)alloc(sizeof(h16) * FFD * DIMM);
  h16* wh_ff2      = (h16*)alloc(sizeof(h16) * DIMM * FFD);
  h16* wh_projout  = (h16*)alloc(sizeof(h16) * 16 * DIMM);   // padded 8 -> 16 rows
  h16* wh_mlp1     = (h16*)alloc(sizeof(h16) * DIMM * LL);
  h16* wh_mlp2     = (h16*)alloc(sizeof(h16) * DIMM * DIMM);
  h16* wh_mlp3     = (h16*)alloc(sizeof(h16) * 16 * DIMM);   // padded 8 -> 16 rows

  float* xf    = (float*)alloc(sizeof(float) * ML * DIMM);
  h16*   xh    = (h16*)  alloc(sizeof(h16)   * ML * DIMM);
  h16*   qkvh  = (h16*)  alloc(sizeof(h16)   * ML * 3 * DIMM);
  h16*   ctxh  = (h16*)  alloc(sizeof(h16)   * ML * DIMM);
  float* r1f   = (float*)alloc(sizeof(float) * ML * DIMM);
  float* h1f   = (float*)alloc(sizeof(float) * ML * DIMM);
  h16*   h1h   = (h16*)  alloc(sizeof(h16)   * ML * DIMM);
  h16*   ffh   = (h16*)  alloc(sizeof(h16)   * ML * FFD);
  float* r2f   = (float*)alloc(sizeof(float) * ML * DIMM);
  h16*   h2h   = (h16*)  alloc(sizeof(h16)   * ML * DIMM);
  h16*   accYh = (h16*)  alloc(sizeof(h16)   * MP * LL);
  h16*   p0    = (h16*)  alloc(sizeof(h16)   * MP * DIMM);
  h16*   p1    = (h16*)  alloc(sizeof(h16)   * MP * DIMM);
  float* phi   = (float*)alloc(sizeof(float) * MP * MODES);
  float* amax  = (float*)alloc(sizeof(float) * BB * MODES);

  auto conv = [&](const float* src, h16* dst, int n, int npad) {
    k_f32_to_f16<<<(npad + 255) / 256, 256, 0, stream>>>(src, dst, n, npad);
  };
  // heavy path: 128x64 tiles per block, TDM/LDS-staged weights (N%64==0, K<=256)
  auto gemm_lds = [&](const h16* A, const h16* W, const float* bias, const float* resid,
                      float* oF, h16* oH, int M, int N, int K, int act) {
    dim3 g(M / 128, N / 64);
    k_gemm_lds<<<g, 256, 0, stream>>>(A, W, bias, resid, oF, oH, M, N, K, act);
  };
  // generic path: one wave per 16x16 tile (mlp1 K=1024; N=8 padded to 16)
  auto gemm = [&](const h16* A, const h16* W, const float* bias, const float* resid,
                  float* oF, h16* oH, int M, int N, int Np, int K, int act) {
    int tiles  = (M >> 4) * (Np >> 4);
    int blocks = (tiles + 3) / 4;  // 4 waves / block (128 threads, wave32)
    k_gemm<<<blocks, 128, 0, stream>>>(A, W, bias, resid, oF, oH, M, N, Np, K, act);
  };

  // weight + input conversions to f16 (tiny-N weights zero-padded to 16 rows)
  conv(attn_in_w,  wh_attn_in,  3 * DIMM * DIMM, 3 * DIMM * DIMM);
  conv(attn_out_w, wh_attn_out, DIMM * DIMM,     DIMM * DIMM);
  conv(ff1_w,      wh_ff1,      FFD * DIMM,      FFD * DIMM);
  conv(ff2_w,      wh_ff2,      DIMM * FFD,      DIMM * FFD);
  conv(proj_out_w, wh_projout,  MODES * DIMM,    16 * DIMM);
  conv(mlp1_w,     wh_mlp1,     DIMM * LL,       DIMM * LL);
  conv(mlp2_w,     wh_mlp2,     DIMM * DIMM,     DIMM * DIMM);
  conv(mlp3_w,     wh_mlp3,     MODES * DIMM,    16 * DIMM);
  conv(accY,       accYh,       MP * LL,         MP * LL);

  // ---- q path ----
  k_proj_in<<<(BB * LL * DIMM + 255) / 256, 256, 0, stream>>>(accY, proj_in_w, proj_in_b, xf, xh);
  gemm_lds(xh, wh_attn_in, attn_in_b, nullptr, nullptr, qkvh, ML, 3 * DIMM, DIMM, 0);
  {
    int waves  = BB * HH * (LL / 16);           // 8192
    int blocks = (waves + 3) / 4;
    k_attn<<<blocks, 128, 0, stream>>>(qkvh, ctxh);
  }
  gemm_lds(ctxh, wh_attn_out, attn_out_b, xf, r1f, nullptr, ML, DIMM, DIMM, 0);
  k_layernorm<<<(ML + 7) / 8, 256, 0, stream>>>(r1f, ln1_g, ln1_b, h1f, h1h, ML);
  gemm_lds(h1h, wh_ff1, ff1_b, nullptr, nullptr, ffh, ML, FFD, DIMM, 1);       // relu
  gemm_lds(ffh, wh_ff2, ff2_b, h1f, r2f, nullptr, ML, DIMM, FFD, 0);
  k_layernorm<<<(ML + 7) / 8, 256, 0, stream>>>(r2f, ln2_g, ln2_b, nullptr, h2h, ML);
  gemm(h2h, wh_projout, proj_out_b, nullptr, q_out, nullptr, ML, MODES, 16, DIMM, 0);

  // ---- phi path ----
  gemm(accYh, wh_mlp1, mlp1_b, nullptr, nullptr, p0, MP, DIMM, DIMM, LL, 2);   // tanh
  gemm_lds(p0, wh_mlp2, mlp2_b, nullptr, nullptr, p1, MP, DIMM, DIMM, 2);
  gemm_lds(p1, wh_mlp2, mlp2_b, nullptr, nullptr, p0, MP, DIMM, DIMM, 2);
  gemm(p0, wh_mlp3, mlp3_b, nullptr, phi, nullptr, MP, MODES, 16, DIMM, 0);
  k_absmax<<<(BB * MODES + 3) / 4, 128, 0, stream>>>(phi, amax);
  k_norm<<<(MP * MODES + 255) / 256, 256, 0, stream>>>(phi, amax, phi_out);
}
